// MultiheadAttention_48043504173227
// MI455X (gfx1250) — matrix operand order repaired and checked by reading, not compiled, not run on hardware
//
#include <hip/hip_runtime.h>
#include <hip/hip_bf16.h>

// Sizes fixed by the reference
#define SDIM 1024
#define BDIM 4
#define EDIM 1024
#define HDIM 16
#define DDIM 64
#define NROWS (SDIM * BDIM)        // 4096
#define MAXREL 32
#define NREL (2 * MAXREL + 1)      // 65

typedef __attribute__((ext_vector_type(16))) _Float16 v16h;
typedef __attribute__((ext_vector_type(8)))  _Float16 v8h;
typedef __attribute__((ext_vector_type(8)))  float    v8f;
typedef __attribute__((ext_vector_type(4)))  int      v4i;

union V16H { v16h v; v8h h[2]; };

// ---------------------------------------------------------------------------
// gfx1250 async global->LDS copy (ASYNCcnt path), guarded so the file still
// compiles (falling back to load + ds_store) if the builtins are absent.
// The builtin expects int4 pointers: global (AS1) source, LDS (AS3) dest.
// ---------------------------------------------------------------------------
#if defined(__has_builtin)
#  if __has_builtin(__builtin_amdgcn_global_load_async_to_lds_b128) && \
      __has_builtin(__builtin_amdgcn_s_wait_asynccnt)
#    define USE_ASYNC_LDS 1
#  endif
#endif
#ifndef USE_ASYNC_LDS
#  define USE_ASYNC_LDS 0
#endif

#if USE_ASYNC_LDS
typedef __attribute__((address_space(1))) v4i as1_v4i;
typedef __attribute__((address_space(3))) v4i as3_v4i;
#endif

__device__ __forceinline__ void stage_b128(void* lds_dst, const void* gsrc) {
#if USE_ASYNC_LDS
  __builtin_amdgcn_global_load_async_to_lds_b128(
      (as1_v4i*)gsrc, (as3_v4i*)lds_dst, 0, 0);
#else
  *(v8h*)lds_dst = *(const v8h*)gsrc;
#endif
}

__device__ __forceinline__ void stage_wait() {
#if USE_ASYNC_LDS
  __builtin_amdgcn_s_wait_asynccnt(0);
#endif
}

// ---------------------------------------------------------------------------
// Elementwise f32 -> f16 conversion
// ---------------------------------------------------------------------------
__global__ void cvt_f16_kernel(const float* __restrict__ src,
                               _Float16* __restrict__ dst, int n) {
  int i = blockIdx.x * blockDim.x + threadIdx.x;
  if (i < n) dst[i] = (_Float16)src[i];
}

// ---------------------------------------------------------------------------
// WMMA GEMM: C[n][e] = sum_k A[n][k] * W[e][k]  (+bias, epilogue modes)
//   A: NROWS x EDIM f16 row-major, W: EDIM x EDIM f16 row-major
//   mode 0: f32 out (n,e) row-major, +bias            (final projection)
//   mode 1: (+bias)*scale, RoPE, f16 out (B,H,S,D)    (Q with scale, K)
//   mode 2: +bias, f16 out (B,H,S,D)                  (V)
// Block: 256 threads = 8 waves; tile M=128 (16 rows/wave), N=64, K-step 32.
// ---------------------------------------------------------------------------
__global__ __launch_bounds__(256)
void gemm_wmma_kernel(const _Float16* __restrict__ A,
                      const _Float16* __restrict__ W,
                      const float* __restrict__ bias,
                      void* __restrict__ out,
                      int mode, float scale) {
  __shared__ __align__(16) _Float16 wLds[64][40];   // 64 cols x 32 K (pad 40)

  const int wave = threadIdx.x >> 5;
  const int lane = threadIdx.x & 31;
  const int l15  = lane & 15;
  const int hi   = lane >> 4;
  const int Mbase = blockIdx.x * 128 + wave * 16;
  const int Nbase = blockIdx.y * 64;

  v8f zero = {};
  v8f acc[4];
  #pragma unroll
  for (int i = 0; i < 4; ++i) acc[i] = zero;

  const int arow = Mbase + l15;
  const _Float16* arp = A + (size_t)arow * EDIM;

  for (int koff = 0; koff < EDIM; koff += 32) {
    // Stage W tile: rows Nbase..Nbase+63, cols koff..koff+31 (2048 halves)
    {
      int t = threadIdx.x;               // 256 threads * 16B = 2048 halves
      int row = t >> 2, ch = (t & 3) * 8;
      stage_b128(&wLds[row][ch],
                 W + (size_t)(Nbase + row) * EDIM + koff + ch);
      stage_wait();
    }
    __syncthreads();

    if (koff + 32 < EDIM)
      __builtin_prefetch(arp + koff + 32, 0, 1);

    // A fragment: per-lane K halves [off..off+7] and [off+16..off+23]
    V16H a;
    const _Float16* ap = arp + koff + hi * 8;
    a.h[0] = *(const v8h*)(ap);
    a.h[1] = *(const v8h*)(ap + 16);

    #pragma unroll
    for (int nt = 0; nt < 4; ++nt) {
      V16H b;   // B fragment: K halves [8*hi..+7] and [16+8*hi..+7]
      const _Float16* bp = &wLds[nt * 16 + l15][hi * 8];
      b.h[0] = *(const v8h*)(bp);
      b.h[1] = *(const v8h*)(bp + 16);
      acc[nt] = __builtin_amdgcn_wmma_f32_16x16x32_f16(
          false, a.v, false, b.v, (short)0, acc[nt], false, false);
    }
    __syncthreads();
  }

  // Epilogue. C layout: VGPR j -> row j + 8*hi, col = lane&15.
  #pragma unroll
  for (int nt = 0; nt < 4; ++nt) {
    #pragma unroll
    for (int j = 0; j < 8; ++j) {
      int m = Mbase + j + 8 * hi;             // output row n (= s*B + b)
      int e = Nbase + nt * 16 + l15;          // output col
      float val = acc[nt][j] + bias[e];
      if (mode == 0) {
        ((float*)out)[(size_t)m * EDIM + e] = val;
      } else {
        int s  = m >> 2;                      // BDIM == 4
        int bb = m & 3;
        int hh = e >> 6;                      // DDIM == 64
        int dd = e & 63;
        float o;
        if (mode == 1) {
          val *= scale;
          // rotate_every_two partner sits in the adjacent lane (col^1)
          float partner = __shfl_xor(val, 1, 32);
          float fi  = (float)(dd & 31);
          float inv = __powf(10000.0f, -fi * (1.0f / 32.0f));
          float ang = (float)s * inv;
          float c = __cosf(ang), sn = __sinf(ang);
          float rot = (dd & 1) ? partner : -partner;
          o = val * c + rot * sn;
        } else {
          o = val;
        }
        ((_Float16*)out)[(((size_t)bb * HDIM + hh) * SDIM + s) * DDIM + dd] =
            (_Float16)o;
      }
    }
  }
}

// ---------------------------------------------------------------------------
// qrel[(b*H+h)*S + s][r] = dot(Q[b,h,s,:], rel_emb[r,:]),  r in [0,65)
// ---------------------------------------------------------------------------
__global__ __launch_bounds__(96)
void qrel_kernel(const _Float16* __restrict__ Qh,
                 const float* __restrict__ rel,
                 float* __restrict__ qrel) {
  __shared__ float qrow[DDIM];
  int row = blockIdx.x;
  int t = threadIdx.x;
  if (t < DDIM) qrow[t] = (float)Qh[(size_t)row * DDIM + t];
  __syncthreads();
  if (t < NREL) {
    const float* rp = rel + (size_t)t * DDIM;
    float acc = 0.f;
    #pragma unroll
    for (int d = 0; d < DDIM; ++d) acc += qrow[d] * rp[d];
    qrel[(size_t)row * NREL + t] = acc;
  }
}

// ---------------------------------------------------------------------------
// Fused flash attention per (b,h): online softmax, WMMA for QK^T and PV.
// Block: 128 threads = 4 waves; each wave owns 16 q rows of a 64-row q tile.
// ---------------------------------------------------------------------------
__global__ __launch_bounds__(128)
void attn_wmma_kernel(const _Float16* __restrict__ Qh,
                      const _Float16* __restrict__ Kh,
                      const _Float16* __restrict__ Vh,
                      const float* __restrict__ qrel,
                      const float* __restrict__ mask,
                      _Float16* __restrict__ Ah) {
  __shared__ __align__(16) _Float16 kT[64][72];       // [key][d]
  __shared__ __align__(16) _Float16 vT[64][72];       // [d][key] (transposed)
  __shared__ __align__(16) _Float16 pL[4][16][72];    // per-wave P staging

  const int bh = blockIdx.y;                  // b*H + h
  const int b  = bh >> 4;                     // HDIM == 16
  const int h  = bh & 15;
  const int qtile = blockIdx.x * 64;
  const int wave = threadIdx.x >> 5;
  const int lane = threadIdx.x & 31;
  const int l15 = lane & 15, hi = lane >> 4;
  const size_t bhS = (size_t)bh * SDIM;
  const int qrow0 = qtile + wave * 16;

  // Q fragments (invariant across k tiles): K-dim = D = 64 -> 2 k-steps
  V16H aq[2];
  {
    const _Float16* qp = Qh + (bhS + qrow0 + l15) * DDIM;
    #pragma unroll
    for (int ks = 0; ks < 2; ++ks) {
      aq[ks].h[0] = *(const v8h*)(qp + ks * 32 + hi * 8);
      aq[ks].h[1] = *(const v8h*)(qp + ks * 32 + hi * 8 + 16);
    }
  }

  v8f zero = {};
  v8f O[4];
  #pragma unroll
  for (int i = 0; i < 4; ++i) O[i] = zero;
  float mrow[8], lrow[8];
  #pragma unroll
  for (int j = 0; j < 8; ++j) { mrow[j] = -3.0e38f; lrow[j] = 0.f; }

  for (int kt = 0; kt < SDIM; kt += 64) {
    // Stage K tile async (row-major) and V tile (manual transpose)
    {
      int t = threadIdx.x;
      #pragma unroll
      for (int p = 0; p < 4; ++p) {
        int idx = p * 128 + t;                // 512 chunks of 8 halves
        int row = idx >> 3, ch = (idx & 7) * 8;
        stage_b128(&kT[row][ch],
                   Kh + (bhS + kt + row) * DDIM + ch);
        v8h vv = *(const v8h*)(Vh + (bhS + kt + row) * DDIM + ch);
        #pragma unroll
        for (int q = 0; q < 8; ++q) vT[ch + q][row] = vv[q];
      }
      stage_wait();
    }
    __syncthreads();

    // Scores: S = Q * K^T  (64 keys, K-dim = 64)
    v8f sc[4];
    #pragma unroll
    for (int i = 0; i < 4; ++i) sc[i] = zero;
    #pragma unroll
    for (int ks = 0; ks < 2; ++ks) {
      #pragma unroll
      for (int nt = 0; nt < 4; ++nt) {
        V16H bf;    // lane n = key; d halves [8*hi..+7] and [16+8*hi..+7]
        const _Float16* bp = &kT[nt * 16 + l15][ks * 32 + hi * 8];
        bf.h[0] = *(const v8h*)(bp);
        bf.h[1] = *(const v8h*)(bp + 16);
        sc[nt] = __builtin_amdgcn_wmma_f32_16x16x32_f16(
            false, aq[ks].v, false, bf.v, (short)0, sc[nt], false, false);
      }
    }

    // rel + region mask, online softmax update, stage P (f16) in LDS
    #pragma unroll
    for (int j = 0; j < 8; ++j) {
      int qg = qrow0 + j + 8 * hi;
      float sv[4];
      #pragma unroll
      for (int nt = 0; nt < 4; ++nt) {
        int kg = kt + nt * 16 + l15;
        int dist = kg - qg;
        dist = dist < -MAXREL ? -MAXREL : (dist > MAXREL ? MAXREL : dist);
        sv[nt] = sc[nt][j]
               + qrel[((size_t)bh * SDIM + qg) * NREL + (dist + MAXREL)]
               + mask[((size_t)b * SDIM + qg) * SDIM + kg];
      }
      float tmax = fmaxf(fmaxf(sv[0], sv[1]), fmaxf(sv[2], sv[3]));
      #pragma unroll
      for (int o = 8; o >= 1; o >>= 1)
        tmax = fmaxf(tmax, __shfl_xor(tmax, o, 32));
      float mnew = fmaxf(mrow[j], tmax);
      float corr = __expf(mrow[j] - mnew);
      float rsum = 0.f;
      #pragma unroll
      for (int nt = 0; nt < 4; ++nt) {
        float p = __expf(sv[nt] - mnew);
        rsum += p;
        pL[wave][j + 8 * hi][nt * 16 + l15] = (_Float16)p;
      }
      #pragma unroll
      for (int o = 8; o >= 1; o >>= 1)
        rsum += __shfl_xor(rsum, o, 32);
      lrow[j] = lrow[j] * corr + rsum;
      mrow[j] = mnew;
      #pragma unroll
      for (int od = 0; od < 4; ++od) O[od][j] *= corr;
    }

    // O += P * V   (K-dim = 64 keys -> 2 k-steps)
    #pragma unroll
    for (int ks = 0; ks < 2; ++ks) {
      V16H ap;   // A-layout read of staged P
      const _Float16* pp = &pL[wave][l15][ks * 32 + hi * 8];
      ap.h[0] = *(const v8h*)(pp);
      ap.h[1] = *(const v8h*)(pp + 16);
      #pragma unroll
      for (int od = 0; od < 4; ++od) {
        V16H bf;   // lane n = d col; key halves [8*hi..+7] and [16+8*hi..+7]
        const _Float16* vp = &vT[od * 16 + l15][ks * 32 + hi * 8];
        bf.h[0] = *(const v8h*)(vp);
        bf.h[1] = *(const v8h*)(vp + 16);
        O[od] = __builtin_amdgcn_wmma_f32_16x16x32_f16(
            false, ap.v, false, bf.v, (short)0, O[od], false, false);
      }
    }
    __syncthreads();
  }

  // Normalize and emit attn in (s, b, e) layout as f16 for the final GEMM
  #pragma unroll
  for (int od = 0; od < 4; ++od) {
    #pragma unroll
    for (int j = 0; j < 8; ++j) {
      int qg = qrow0 + j + 8 * hi;
      float o = O[od][j] / lrow[j];
      int e = h * 64 + od * 16 + l15;
      Ah[(size_t)(qg * BDIM + b) * EDIM + e] = (_Float16)o;
    }
  }
}

// ---------------------------------------------------------------------------
extern "C" void kernel_launch(void* const* d_in, const int* in_sizes, int n_in,
                              void* d_out, int out_size, void* d_ws,
                              size_t ws_size, hipStream_t stream) {
  const float* query = (const float*)d_in[0];
  const float* mask  = (const float*)d_in[1];
  const float* Wq = (const float*)d_in[2];
  const float* bq = (const float*)d_in[3];
  const float* Wk = (const float*)d_in[4];
  const float* bk = (const float*)d_in[5];
  const float* Wv = (const float*)d_in[6];
  const float* bv = (const float*)d_in[7];
  const float* Wo = (const float*)d_in[8];
  const float* bo = (const float*)d_in[9];
  const float* rel = (const float*)d_in[10];
  float* out = (float*)d_out;

  char* ws = (char*)d_ws;
  size_t off = 0;
  auto alloc = [&](size_t bytes) -> void* {
    void* p = ws + off;
    off = (off + bytes + 255) & ~(size_t)255;
    return p;
  };

  _Float16* Xh  = (_Float16*)alloc((size_t)NROWS * EDIM * 2);
  _Float16* Wqh = (_Float16*)alloc((size_t)EDIM * EDIM * 2);
  _Float16* Wkh = (_Float16*)alloc((size_t)EDIM * EDIM * 2);
  _Float16* Wvh = (_Float16*)alloc((size_t)EDIM * EDIM * 2);
  _Float16* Woh = (_Float16*)alloc((size_t)EDIM * EDIM * 2);
  _Float16* Qh  = (_Float16*)alloc((size_t)BDIM * HDIM * SDIM * DDIM * 2);
  _Float16* Kh  = (_Float16*)alloc((size_t)BDIM * HDIM * SDIM * DDIM * 2);
  _Float16* Vh  = (_Float16*)alloc((size_t)BDIM * HDIM * SDIM * DDIM * 2);
  float*    Qr  = (float*)alloc((size_t)BDIM * HDIM * SDIM * NREL * 4);
  _Float16* Ahb = (_Float16*)alloc((size_t)NROWS * EDIM * 2);

  const int nX = NROWS * EDIM;
  const int nW = EDIM * EDIM;
  cvt_f16_kernel<<<(nX + 255) / 256, 256, 0, stream>>>(query, Xh, nX);
  cvt_f16_kernel<<<(nW + 255) / 256, 256, 0, stream>>>(Wq, Wqh, nW);
  cvt_f16_kernel<<<(nW + 255) / 256, 256, 0, stream>>>(Wk, Wkh, nW);
  cvt_f16_kernel<<<(nW + 255) / 256, 256, 0, stream>>>(Wv, Wvh, nW);
  cvt_f16_kernel<<<(nW + 255) / 256, 256, 0, stream>>>(Wo, Woh, nW);

  dim3 ggrid(NROWS / 128, EDIM / 64);
  // Q: +bias, *D^-0.5, RoPE   K: +bias, RoPE   V: +bias
  gemm_wmma_kernel<<<ggrid, 256, 0, stream>>>(Xh, Wqh, bq, Qh, 1, 0.125f);
  gemm_wmma_kernel<<<ggrid, 256, 0, stream>>>(Xh, Wkh, bk, Kh, 1, 1.0f);
  gemm_wmma_kernel<<<ggrid, 256, 0, stream>>>(Xh, Wvh, bv, Vh, 2, 1.0f);

  qrel_kernel<<<BDIM * HDIM * SDIM, 96, 0, stream>>>(Qh, rel, Qr);

  dim3 agrid(SDIM / 64, BDIM * HDIM);
  attn_wmma_kernel<<<agrid, 128, 0, stream>>>(Qh, Kh, Vh, Qr, mask, Ahb);

  // Final projection -> d_out (f32)
  gemm_wmma_kernel<<<ggrid, 256, 0, stream>>>(Ahb, Woh, bo, out, 0, 1.0f);
}
